// RNN_16527034155307
// MI455X (gfx1250) — matrix operand order repaired and checked by reading, not compiled, not run on hardware
//
#include <hip/hip_runtime.h>
#include <hip/hip_bf16.h>

// ---------------------------------------------------------------------------
// Problem constants (from reference): B=64, T=512, IN2=128, H=512, OUT=256, L=2
// ---------------------------------------------------------------------------
#define BB   64
#define TT   512
#define IN2C 128
#define HH   512
#define H2   1024     // 2*H
#define G3   1536     // 3*H
#define OUTN 256
#define BT   (BB * TT)   // 32768

typedef __bf16 bf16;
typedef __attribute__((ext_vector_type(16))) __bf16 v16bf;
typedef __attribute__((ext_vector_type(8)))  __bf16 v8bf;
typedef __attribute__((ext_vector_type(8)))  float  v8f;

// ---------------------------------------------------------------------------
// WMMA helpers (CDNA5: V_WMMA_F32_16X16X32_BF16, D = A(16x32) * B(32x16) + C)
// A-frag per lane: rows m0+(lane&15); K elems at kb+(lane>>4)*8 .. +7 and +16.
// B-frag per lane: W row n0+(lane&15); K elems at kb+(lane>>4)*8 .. +7 and +16 (as A).
// C/D v8f: elem i -> m = m0 + i + 8*(lane>=16), n = n0 + (lane&15).
// ---------------------------------------------------------------------------
__device__ __forceinline__ v16bf frag_cat(v8bf lo, v8bf hi) {
    return __builtin_shufflevector(lo, hi, 0, 1, 2, 3, 4, 5, 6, 7,
                                           8, 9, 10, 11, 12, 13, 14, 15);
}

__device__ __forceinline__ v8f wmma_bf16(v16bf a, v16bf b, v8f c) {
    return __builtin_amdgcn_wmma_f32_16x16x32_bf16(
        /*neg_a=*/false, a, /*neg_b=*/false, b,
        /*c_mod=*/(short)0, c, /*reuse_a=*/false, /*reuse_b=*/false);
}

__device__ __forceinline__ float sigmoidf_fast(float x) {
    return 1.0f / (1.0f + __expf(-x));
}
__device__ __forceinline__ float tanhf_fast(float x) {
    return 2.0f * sigmoidf_fast(2.0f * x) - 1.0f;
}

// XOR swizzle of 8-element (16B) granules within a 512-wide LDS row: kills
// the 16-way bank conflict of same-column A-fragment loads across rows.
__device__ __forceinline__ int swz(int m, int j) {
    return (((j >> 3) ^ (m & 7)) << 3) | (j & 7);
}

// ---------------------------------------------------------------------------
// Kernel 1: fp32 -> bf16 conversion (grid-stride)
// ---------------------------------------------------------------------------
__global__ void cvt_bf16_kernel(const float* __restrict__ s,
                                bf16* __restrict__ d, long long n) {
    long long i  = (long long)blockIdx.x * blockDim.x + threadIdx.x;
    long long st = (long long)gridDim.x * blockDim.x;
    for (; i < n; i += st) d[i] = (bf16)s[i];
}

// ---------------------------------------------------------------------------
// Kernel 2: input projection.
//   x0[t, b, half*H + n] = input[b, t, half*64 : half*64+64] @ xy_w^T + xy_b
// Output time-major (T, B, 2H) bf16. One 16x16 tile per wave, K=64.
// ---------------------------------------------------------------------------
__global__ __launch_bounds__(256) void proj_kernel(
    const bf16* __restrict__ inb,   // (B, T, IN2) bf16
    const bf16* __restrict__ wxy,   // (H, 64) bf16
    const float* __restrict__ bxy,  // (H,)
    bf16* __restrict__ x0)          // (T, B, 2H) bf16
{
    int lane = threadIdx.x & 31;
    int wid  = threadIdx.x >> 5;
    int tile = blockIdx.x * 8 + wid;          // half(2) x mt(2048) x nt(32)
    int nt   = tile & 31;  tile >>= 5;
    int mt   = tile & 2047; tile >>= 11;
    int half = tile;
    int m0 = mt * 16, n0 = nt * 16;

    // out row m -> (t = m/64, b = m%64); input row = b*T + t
    int m = m0 + (lane & 15);
    int t = m >> 6, b = m & 63;
    const bf16* arow = inb + ((size_t)b * TT + t) * IN2C + half * 64;
    int koA = (lane >> 4) * 8;
    int koB = (lane >> 4) * 8;

    v8f acc = {};
#pragma unroll
    for (int k = 0; k < 64; k += 32) {
        v16bf a = frag_cat(*(const v8bf*)(arow + k + koA),
                           *(const v8bf*)(arow + k + koA + 16));
        const bf16* pb = wxy + (size_t)(n0 + (lane & 15)) * 64 + k + koB;
        v16bf bfr = frag_cat(*(const v8bf*)pb, *(const v8bf*)(pb + 16));
        acc = wmma_bf16(a, bfr, acc);
    }
    int n = n0 + (lane & 15);
    float bias = bxy[n];
#pragma unroll
    for (int i = 0; i < 8; i++) {
        int mm = m0 + i + ((lane >> 4) << 3);
        x0[(size_t)mm * H2 + (size_t)half * HH + n] = (bf16)(acc[i] + bias);
    }
}

// ---------------------------------------------------------------------------
// Kernel 3: xg = X @ W_ih^T + b_ih  (M=BT, N=1536, K=1024).
// Output TRANSPOSED as (T, 3H, B) bf16 so that both the epilogue stores here
// and the per-step loads in the scan are 16B-contiguous per lane (the C/D
// fragment holds 8 consecutive m per lane).
// 64x128 tile per 256-thread block, LDS-staged, K-slab = 64.
// ---------------------------------------------------------------------------
__global__ __launch_bounds__(256) void gemm_xg_kernel(
    const bf16* __restrict__ X,     // (BT, 2H) bf16, time-major rows
    const bf16* __restrict__ W,     // (3H, 2H) bf16
    const float* __restrict__ bias, // (3H,)
    bf16* __restrict__ xgt)         // (T, 3H, B) bf16  [transposed]
{
    __shared__ bf16 As[64 * 72];    // 64 x 64, row stride 72 (bank spread)
    __shared__ bf16 Bs[128 * 72];   // 128 x 64

    int lane  = threadIdx.x & 31, wid = threadIdx.x >> 5;
    int mbase = blockIdx.y * 64;              // multiple of 64 => single t
    int nbase = blockIdx.x * 128;
    int wm = wid >> 1, wn = wid & 1;

    int ra = threadIdx.x >> 2, ca = (threadIdx.x & 3) * 16;  // A: 32B/thread
    int rb = threadIdx.x >> 1, cb = (threadIdx.x & 1) * 32;  // B: 64B/thread

    v8f acc[4] = {};

    for (int k0 = 0; k0 < H2; k0 += 64) {
        // cooperative stage into LDS (K-slab of 64)
        const bf16* xp = &X[(size_t)(mbase + ra) * H2 + k0 + ca];
        *(v8bf*)&As[ra * 72 + ca]     = *(const v8bf*)xp;
        *(v8bf*)&As[ra * 72 + ca + 8] = *(const v8bf*)(xp + 8);
        const bf16* wp = &W[(size_t)(nbase + rb) * H2 + k0 + cb];
        *(v8bf*)&Bs[rb * 72 + cb]      = *(const v8bf*)wp;
        *(v8bf*)&Bs[rb * 72 + cb + 8]  = *(const v8bf*)(wp + 8);
        *(v8bf*)&Bs[rb * 72 + cb + 16] = *(const v8bf*)(wp + 16);
        *(v8bf*)&Bs[rb * 72 + cb + 24] = *(const v8bf*)(wp + 24);
        if (k0 + 64 < H2) {   // gfx1250 global_prefetch_b8 on next K slab
            __builtin_prefetch(xp + 64, 0, 1);
            __builtin_prefetch(wp + 64, 0, 1);
        }
        __syncthreads();

        int mA  = wm * 16 + (lane & 15);
        int koA = (lane >> 4) * 8;
        int koB = (lane >> 4) * 8;
#pragma unroll
        for (int kk = 0; kk < 64; kk += 32) {
            v16bf a = frag_cat(*(const v8bf*)&As[mA * 72 + kk + koA],
                               *(const v8bf*)&As[mA * 72 + kk + koA + 16]);
#pragma unroll
            for (int j = 0; j < 4; j++) {
                int rB = wn * 64 + j * 16 + (lane & 15);
                v16bf bfr = frag_cat(*(const v8bf*)&Bs[rB * 72 + kk + koB],
                                     *(const v8bf*)&Bs[rB * 72 + kk + koB + 16]);
                acc[j] = wmma_bf16(a, bfr, acc[j]);
            }
        }
        __syncthreads();
    }

    // epilogue: per lane, 8 consecutive m => one 16B store per tile
    int t  = mbase >> 6;                         // row block = single t
    int b0 = wm * 16 + ((lane >> 4) << 3);       // first of 8 consecutive b
#pragma unroll
    for (int j = 0; j < 4; j++) {
        int n = nbase + wn * 64 + j * 16 + (lane & 15);
        float bv = bias[n];
        v8bf pk;
#pragma unroll
        for (int i = 0; i < 8; i++) pk[i] = (bf16)(acc[j][i] + bv);
        *(v8bf*)&xgt[((size_t)t * G3 + n) * BB + b0] = pk;
    }
}

// ---------------------------------------------------------------------------
// Kernel 4: the sequential GRU scan. grid = 2 (one workgroup per direction),
// 1024 threads = 32 waves. h (64x512 bf16 = 64KB) lives in LDS (swizzled).
// Wave w owns output columns j in [16w, 16w+16) for all 4 m-tiles, 3 gates.
// Per step: hg = h @ W_hh^T (WMMA from LDS A / L2-resident B), gate math
// (xg read as 16B vectors thanks to the (T,3H,B) layout), barrier, h update.
// ---------------------------------------------------------------------------
__global__ __launch_bounds__(1024) void scan_kernel(
    const bf16* __restrict__ xg0,   // (T, 3H, B) dir 0
    const bf16* __restrict__ xg1,   // (T, 3H, B) dir 1
    const bf16* __restrict__ whh,   // (2, 3H, H) bf16
    const float* __restrict__ bhh,  // (2, 3H)
    bf16* __restrict__ Xout)        // (T*B, 2H) bf16
{
    __shared__ bf16 hsm[BB * HH];   // exactly 64 KB

    int d = blockIdx.x;
    const bf16*  xg = d ? xg1 : xg0;
    const bf16*  W  = whh + (size_t)d * G3 * HH;
    const float* bh = bhh + (size_t)d * G3;

    int lane = threadIdx.x & 31, wv = threadIdx.x >> 5;
    for (int i = threadIdx.x; i < BB * HH; i += 1024) hsm[i] = (bf16)0.0f;
    __syncthreads();

    int n0j   = wv * 16;
    int mlane = lane & 15;
    int nl    = n0j + mlane;
    int hilo  = lane >> 4;
    int koA   = hilo * 8, koB = hilo * 8;
    float br = bh[nl], bz = bh[HH + nl], bn = bh[2 * HH + nl];

    for (int ti = 0; ti < TT; ++ti) {
        int t = d ? (TT - 1 - ti) : ti;
        float hstash[4][8];

#pragma unroll
        for (int s = 0; s < 4; s++) {
            int m0 = s * 16;
            int mA = m0 + mlane;
            int b0 = m0 + hilo * 8;            // 8 consecutive b per lane
            v8f accg[3];
#pragma unroll
            for (int g = 0; g < 3; g++) {
                v8f acc = {};
                for (int k16 = 0; k16 < 16; k16++) {
                    int kb = k16 * 32;
                    int c0 = kb + koA;
                    v16bf a = frag_cat(
                        *(const v8bf*)&hsm[mA * HH + swz(mA, c0)],
                        *(const v8bf*)&hsm[mA * HH + swz(mA, c0 + 16)]);
                    const bf16* pb =
                        &W[(size_t)(g * HH + n0j + mlane) * HH + kb + koB];
                    v16bf bfr = frag_cat(*(const v8bf*)pb,
                                         *(const v8bf*)(pb + 16));
                    acc = wmma_bf16(a, bfr, acc);
                }
                accg[g] = acc;
            }
            // gate pre-activations: one 16B load per gate (vs 8 scalars)
            size_t xbase = (size_t)t * G3;
            v8bf xrv = *(const v8bf*)&xg[(xbase + nl) * BB + b0];
            v8bf xzv = *(const v8bf*)&xg[(xbase + HH + nl) * BB + b0];
            v8bf xnv = *(const v8bf*)&xg[(xbase + 2 * HH + nl) * BB + b0];
#pragma unroll
            for (int i = 0; i < 8; i++) {
                int m = b0 + i;
                float r  = sigmoidf_fast((float)xrv[i] + accg[0][i] + br);
                float z  = sigmoidf_fast((float)xzv[i] + accg[1][i] + bz);
                float nn = tanhf_fast((float)xnv[i] + r * (accg[2][i] + bn));
                float hp = (float)hsm[m * HH + swz(m, nl)];
                hstash[s][i] = (1.0f - z) * nn + z * hp;
            }
        }
        __syncthreads();   // all reads of h done before overwriting
#pragma unroll
        for (int s = 0; s < 4; s++) {
#pragma unroll
            for (int i = 0; i < 8; i++) {
                int m = s * 16 + i + hilo * 8;
                bf16 hv = (bf16)hstash[s][i];
                hsm[m * HH + swz(m, nl)] = hv;
                Xout[((size_t)t * BB + m) * H2 + (size_t)d * HH + nl] = hv;
            }
        }
        __syncthreads();   // h ready for next step
    }
}

// ---------------------------------------------------------------------------
// Kernel 5: readout. out = x_last(64,1024) @ ro_w^T(256,1024) + ro_b, fp32.
// 64 tiles of 16x16, one per wave, grid = 8 x 256 threads.
// ---------------------------------------------------------------------------
__global__ __launch_bounds__(256) void readout_kernel(
    const bf16* __restrict__ Xf,    // (T*B, 2H) bf16 (final layer output)
    const bf16* __restrict__ Wro,   // (OUT, 2H) bf16
    const float* __restrict__ bro,  // (OUT,)
    float* __restrict__ out)        // (B, OUT)
{
    int lane = threadIdx.x & 31, wv = threadIdx.x >> 5;
    int tile = blockIdx.x * 8 + wv;          // mt(4) x nt(16)
    int mt = tile >> 4, nt = tile & 15;
    int m0 = mt * 16, n0 = nt * 16;
    int mlane = lane & 15;
    int koA = (lane >> 4) * 8, koB = (lane >> 4) * 8;

    const bf16* abase = Xf + ((size_t)(TT - 1) * BB) * H2;  // t = T-1 slab
    v8f acc = {};
    for (int kb = 0; kb < H2; kb += 32) {
        const bf16* pa = abase + (size_t)(m0 + mlane) * H2 + kb + koA;
        v16bf a = frag_cat(*(const v8bf*)pa, *(const v8bf*)(pa + 16));
        const bf16* pb = Wro + (size_t)(n0 + mlane) * H2 + kb + koB;
        v16bf bfr = frag_cat(*(const v8bf*)pb, *(const v8bf*)(pb + 16));
        acc = wmma_bf16(a, bfr, acc);
    }
    int n = n0 + mlane;
    float bv = bro[n];
#pragma unroll
    for (int i = 0; i < 8; i++) {
        int m = m0 + i + (lane >> 4) * 8;
        out[(size_t)m * OUTN + n] = acc[i] + bv;
    }
}

// ---------------------------------------------------------------------------
// Host orchestration
// ---------------------------------------------------------------------------
extern "C" void kernel_launch(void* const* d_in, const int* in_sizes, int n_in,
                              void* d_out, int out_size, void* d_ws,
                              size_t ws_size, hipStream_t stream) {
    const float* input = (const float*)d_in[0];   // (B,T,IN2)
    const float* xy_w  = (const float*)d_in[1];   // (H,64)
    const float* xy_b  = (const float*)d_in[2];   // (H,)
    const float* w_ih  = (const float*)d_in[3];   // (L,2,3H,2H)
    const float* w_hh  = (const float*)d_in[4];   // (L,2,3H,H)
    const float* b_ih  = (const float*)d_in[5];   // (L,2,3H)
    const float* b_hh  = (const float*)d_in[6];   // (L,2,3H)
    const float* ro_w  = (const float*)d_in[7];   // (OUT,2H)
    const float* ro_b  = (const float*)d_in[8];   // (OUT,)
    (void)in_sizes; (void)n_in; (void)out_size; (void)ws_size;

    char*  ws  = (char*)d_ws;
    size_t off = 0;
    auto carve = [&](size_t bytes) {
        char* p = ws + off;
        off += (bytes + 255) & ~(size_t)255;
        return p;
    };
    bf16* in_bf  = (bf16*)carve((size_t)BT * IN2C * 2);
    bf16* xyw_bf = (bf16*)carve((size_t)HH * 64 * 2);
    bf16* wih_bf = (bf16*)carve((size_t)2 * 2 * G3 * H2 * 2);
    bf16* whh_bf = (bf16*)carve((size_t)2 * 2 * G3 * HH * 2);
    bf16* row_bf = (bf16*)carve((size_t)OUTN * H2 * 2);
    bf16* xA     = (bf16*)carve((size_t)BT * H2 * 2);   // layer ping
    bf16* xB     = (bf16*)carve((size_t)BT * H2 * 2);   // layer pong
    bf16* xg0    = (bf16*)carve((size_t)BT * G3 * 2);   // dir-0 preacts (T,3H,B)
    bf16* xg1    = (bf16*)carve((size_t)BT * G3 * 2);   // dir-1 preacts (T,3H,B)

    // --- bf16 conversion of all WMMA operands ---
    cvt_bf16_kernel<<<2048, 256, 0, stream>>>(input, in_bf, (long long)BT * IN2C);
    cvt_bf16_kernel<<<64,   256, 0, stream>>>(xy_w, xyw_bf, (long long)HH * 64);
    cvt_bf16_kernel<<<2048, 256, 0, stream>>>(w_ih, wih_bf, (long long)2 * 2 * G3 * H2);
    cvt_bf16_kernel<<<2048, 256, 0, stream>>>(w_hh, whh_bf, (long long)2 * 2 * G3 * HH);
    cvt_bf16_kernel<<<512,  256, 0, stream>>>(ro_w, row_bf, (long long)OUTN * H2);

    // --- input projection: x0 = [xp ; yp] time-major (T,B,2H) ---
    proj_kernel<<<16384, 256, 0, stream>>>(in_bf, xyw_bf, xy_b, xA);

    // --- layers ---
    for (int l = 0; l < 2; ++l) {
        const bf16* Xin = (l == 0) ? xA : xB;
        bf16*       Xo  = (l == 0) ? xB : xA;
        for (int d = 0; d < 2; ++d) {
            const bf16*  Wl = wih_bf + (size_t)(l * 2 + d) * G3 * H2;
            const float* bl = b_ih + (size_t)(l * 2 + d) * G3;
            bf16* xgd = d ? xg1 : xg0;
            gemm_xg_kernel<<<dim3(G3 / 128, BT / 64), 256, 0, stream>>>(
                Xin, Wl, bl, xgd);
        }
        scan_kernel<<<2, 1024, 0, stream>>>(
            xg0, xg1,
            whh_bf + (size_t)l * 2 * G3 * HH,
            b_hh + (size_t)l * 2 * G3,
            Xo);
    }

    // --- readout from final layer output (in xA after l=1) ---
    readout_kernel<<<8, 256, 0, stream>>>(xA, row_bf, ro_b, (float*)d_out);
}
